// RWKV_Tmix_x070_33724083208453
// MI455X (gfx1250) — matrix operand order repaired and checked by reading, not compiled, not run on hardware
//
#include <hip/hip_runtime.h>

// ---------------- problem constants (RWKV-7 Tmix, layer 1/24) ----------------
#define Bv   2
#define Tv   1024
#define Cv   2048
#define Hh   32
#define Nh   64
#define BT   2048              // B*T
#define BTC  4194304LL         // B*T*C
#define EPSGN 6.4e-4f          // 1e-5 * 64
#define RESCALE_THR 179.2f     // 256 * 0.7

typedef __attribute__((ext_vector_type(8)))  __bf16 v8bf;
typedef __attribute__((ext_vector_type(16))) __bf16 v16bf;
typedef __attribute__((ext_vector_type(8)))  float  v8f;

// ---------------- device helpers ----------------
__device__ __forceinline__ __bf16 f2bf(float f) {
  union { float f; unsigned u; } in; in.f = f;
  unsigned lsb = (in.u >> 16) & 1u;
  unsigned r = in.u + 0x7fffu + lsb;          // round-to-nearest-even
  union { unsigned short s; __bf16 b; } out;
  out.s = (unsigned short)(r >> 16);
  return out.b;
}
__device__ __forceinline__ float sigmoidf_(float x) { return 1.f / (1.f + expf(-x)); }
__device__ __forceinline__ float softplusf_(float x) { return (x > 20.f) ? x : log1pf(expf(x)); }

// ---------------- bf16 WMMA GEMM with async global->LDS double buffering ----------------
//  C[M,N] = A[M,K] @ Bt[N,K]^T
//  block 256 threads (8 waves), tile 128x64, K-step 32; per wave 32x32 = 2x2 wmma tiles.
//  A tile 128x32 bf16 and B tile 64x32 bf16 staged in LDS with 80B row pitch
//  (20-bank stride -> conflict-free ds_load_b128 for 16-lane column reads).
//  Staging uses GLOBAL_LOAD_ASYNC_TO_LDS_B128 + s_wait_asynccnt (ASYNCcnt pipeline):
//  exactly 3 async ops per thread per tile, so waiting asynccnt<=3 after issuing the
//  next tile retires the previous one.
//  Requires M%128==0, K%32==0.  Any N (B rows clamped; stores guarded).
#define APITCH 40   // bf16 elements per LDS row (80 bytes)

__global__ __launch_bounds__(256) void gemm_bf16_kernel(
    const __bf16* __restrict__ A, const __bf16* __restrict__ Bt,
    float* __restrict__ C, int M, int N, int K) {
  __shared__ __align__(16) __bf16 sA[2][128 * APITCH];
  __shared__ __align__(16) __bf16 sB[2][64 * APITCH];

  const int tid  = threadIdx.x;
  const int lane = tid & 31;
  const int wid  = tid >> 5;
  const int half = lane >> 4;   // lane half (selects K sub-ranges)
  const int l16  = lane & 15;
  const int bm = blockIdx.y * 128;
  const int bn = blockIdx.x * 64;
  const int wmr = (wid & 3) * 32;    // wave row offset inside A tile
  const int wnr = (wid >> 2) * 32;   // wave col offset inside B tile

  // staging work split: A = 32B/thread (2 x b128), B = 16B/thread (1 x b128)
  const int arow = tid >> 1, ahalf = tid & 1;
  const int brow = tid >> 2, bq = tid & 3;
  const int brow_g = (bn + brow < N) ? (bn + brow) : (N - 1);   // clamp (dup row, never stored)

  const __bf16* gA = A + (long long)(bm + arow) * K + ahalf * 16;
  const __bf16* gB = Bt + (long long)brow_g * K + bq * 8;
  // low 32 bits of a flat pointer into LDS are the LDS byte offset
  const unsigned ldsA[2] = {
      (unsigned)(size_t)&sA[0][arow * APITCH + ahalf * 16],
      (unsigned)(size_t)&sA[1][arow * APITCH + ahalf * 16] };
  const unsigned ldsB[2] = {
      (unsigned)(size_t)&sB[0][brow * APITCH + bq * 8],
      (unsigned)(size_t)&sB[1][brow * APITCH + bq * 8] };

  auto issue_tile = [&](int kt, int buf) {
    const __bf16* a = gA + kt * 32;
    const __bf16* b = gB + kt * 32;
    // INST_OFFSET is added to BOTH the LDS and the global address (ISA 08 §4.4),
    // so one address pair serves both 16B halves of the A slice.
    asm volatile("global_load_async_to_lds_b128 %0, %1, off"
                 :: "v"(ldsA[buf]), "v"(a) : "memory");
    asm volatile("global_load_async_to_lds_b128 %0, %1, off offset:16"
                 :: "v"(ldsA[buf]), "v"(a) : "memory");
    asm volatile("global_load_async_to_lds_b128 %0, %1, off"
                 :: "v"(ldsB[buf]), "v"(b) : "memory");
  };

  v8f acc00 = {}, acc01 = {}, acc10 = {}, acc11 = {};
  const int nk = K >> 5;

  issue_tile(0, 0);
  for (int i = 0; i < nk; ++i) {
    const int buf = i & 1;
    if (i + 1 < nk) {
      issue_tile(i + 1, buf ^ 1);
      asm volatile("s_wait_asynccnt 0x3" ::: "memory");   // previous tile landed
    } else {
      asm volatile("s_wait_asynccnt 0x0" ::: "memory");
    }
    __syncthreads();

    const __bf16* pa = &sA[buf][0];
    const __bf16* pb = &sB[buf][0];
    const int ar0 = wmr + l16, ar1 = wmr + 16 + l16;
    const int br0 = wnr + l16, br1 = wnr + 16 + l16;
    // A frag: lane<16 -> K {0..7,16..23}, lane>=16 -> K {8..15,24..31}
    v8bf a0lo = *(const v8bf*)(pa + ar0 * APITCH + half * 8);
    v8bf a0hi = *(const v8bf*)(pa + ar0 * APITCH + 16 + half * 8);
    v8bf a1lo = *(const v8bf*)(pa + ar1 * APITCH + half * 8);
    v8bf a1hi = *(const v8bf*)(pa + ar1 * APITCH + 16 + half * 8);
    // B frag: lane = column, lane<16 -> K {0..7,16..23}, lane>=16 -> K {8..15,24..31}
    v8bf b0lo = *(const v8bf*)(pb + br0 * APITCH + half * 8);
    v8bf b0hi = *(const v8bf*)(pb + br0 * APITCH + 16 + half * 8);
    v8bf b1lo = *(const v8bf*)(pb + br1 * APITCH + half * 8);
    v8bf b1hi = *(const v8bf*)(pb + br1 * APITCH + 16 + half * 8);
    v16bf af0 = __builtin_shufflevector(a0lo, a0hi, 0,1,2,3,4,5,6,7,8,9,10,11,12,13,14,15);
    v16bf af1 = __builtin_shufflevector(a1lo, a1hi, 0,1,2,3,4,5,6,7,8,9,10,11,12,13,14,15);
    v16bf bf0 = __builtin_shufflevector(b0lo, b0hi, 0,1,2,3,4,5,6,7,8,9,10,11,12,13,14,15);
    v16bf bf1 = __builtin_shufflevector(b1lo, b1hi, 0,1,2,3,4,5,6,7,8,9,10,11,12,13,14,15);

    acc00 = __builtin_amdgcn_wmma_f32_16x16x32_bf16(false, af0, false, bf0, (short)0, acc00, false, false);
    acc01 = __builtin_amdgcn_wmma_f32_16x16x32_bf16(false, af0, false, bf1, (short)0, acc01, false, false);
    acc10 = __builtin_amdgcn_wmma_f32_16x16x32_bf16(false, af1, false, bf0, (short)0, acc10, false, false);
    acc11 = __builtin_amdgcn_wmma_f32_16x16x32_bf16(false, af1, false, bf1, (short)0, acc11, false, false);

    __syncthreads();   // all waves done reading buf before tile i+2 overwrites it
  }

  // f32 C/D layout: vgpr r -> row r (lanes 0-15) / row r+8 (lanes 16-31), col = lane%16
  const int col0 = bn + wnr + l16, col1 = bn + wnr + 16 + l16;
  const int rbase = half * 8;
  const int wm = bm + wmr;
#pragma unroll
  for (int rr = 0; rr < 8; ++rr) {
    const int row0 = wm + rr + rbase;
    const int row1 = wm + 16 + rr + rbase;
    if (col0 < N) {
      C[(long long)row0 * N + col0] = acc00[rr];
      C[(long long)row1 * N + col0] = acc10[rr];
    }
    if (col1 < N) {
      C[(long long)row0 * N + col1] = acc01[rr];
      C[(long long)row1 * N + col1] = acc11[rr];
    }
  }
}

// ---------------- f32 -> bf16 weight conversion (optional transpose to [N,K]) ----------------
__global__ __launch_bounds__(256) void convert_bf16_kernel(
    const float* __restrict__ src, __bf16* __restrict__ dst,
    int rows, int cols, int transpose) {
  long long idx = blockIdx.x * 256LL + threadIdx.x;
  long long n = (long long)rows * cols;
  if (idx >= n) return;
  int rr = (int)(idx / cols), cc = (int)(idx % cols);
  __bf16 v = f2bf(src[idx]);
  if (transpose) dst[(long long)cc * rows + rr] = v;
  else           dst[idx] = v;
}

// ---------------- token shift + 6 mixes, emit bf16 GEMM inputs ----------------
__global__ __launch_bounds__(256) void prep_mix_kernel(
    const float* __restrict__ x,
    const float* __restrict__ mr, const float* __restrict__ mw,
    const float* __restrict__ mk, const float* __restrict__ mv,
    const float* __restrict__ ma, const float* __restrict__ mg,
    __bf16* __restrict__ xr, __bf16* __restrict__ xw, __bf16* __restrict__ xk,
    __bf16* __restrict__ xv, __bf16* __restrict__ xa, __bf16* __restrict__ xg) {
  long long idx = blockIdx.x * 256LL + threadIdx.x;
  if (idx >= BTC) return;
  const int c = (int)(idx & (Cv - 1));
  const int t = (int)((idx >> 11) & (Tv - 1));   // Cv = 2^11
  const float xc = x[idx];
  const float xp = (t > 0) ? x[idx - Cv] : 0.f;
  const float dx = xp - xc;
  xr[idx] = f2bf(xc + dx * mr[c]);
  xw[idx] = f2bf(xc + dx * mw[c]);
  xk[idx] = f2bf(xc + dx * mk[c]);
  xv[idx] = f2bf(xc + dx * mv[c]);
  xa[idx] = f2bf(xc + dx * ma[c]);
  xg[idx] = f2bf(xc + dx * mg[c]);
}

// ---------------- global max|a| (or max|a-b|) via atomicMax on float bits ----------------
__global__ __launch_bounds__(256) void maxabs_kernel(
    const float* __restrict__ a, const float* __restrict__ bsub,
    long long n, unsigned* __restrict__ out) {
  long long i = blockIdx.x * 256LL + threadIdx.x;
  const long long stride = (long long)gridDim.x * 256LL;
  float m = 0.f;
  for (; i < n; i += stride) {
    float v = a[i];
    if (bsub) v -= bsub[i];
    m = fmaxf(m, fabsf(v));
  }
  for (int off = 16; off >= 1; off >>= 1) m = fmaxf(m, __shfl_xor(m, off, 32));
  __shared__ float sm[8];
  const int lane = threadIdx.x & 31, wid = threadIdx.x >> 5;
  if (lane == 0) sm[wid] = m;
  __syncthreads();
  if (threadIdx.x == 0) {
    float mm = sm[0];
    for (int wq = 1; wq < 8; ++wq) mm = fmaxf(mm, sm[wq]);
    atomicMax(out, __float_as_uint(mm));
  }
}

// ---------------- f32 -> bf16 with optional activation / scale ----------------
// mode: 0=copy  1=tanh  2=sigmoid  3=divide by max(scale,1)
__global__ __launch_bounds__(256) void act_bf16_kernel(
    const float* __restrict__ in, __bf16* __restrict__ out, long long n,
    int mode, const unsigned* __restrict__ scale_bits) {
  long long idx = blockIdx.x * 256LL + threadIdx.x;
  if (idx >= n) return;
  float v = in[idx];
  if (mode == 1)      v = tanhf(v);
  else if (mode == 2) v = sigmoidf_(v);
  else if (mode == 3) v = v / fmaxf(__uint_as_float(*scale_bits), 1.f);
  out[idx] = f2bf(v);
}

// ---------------- fused elementwise stage before the scan ----------------
// One wave per (b,t,h) row of 64 channels (2 per lane): decay, a, k_final,
// value-residual gate, and L2-normalized kk.
__global__ __launch_bounds__(256) void post_kernel(
    const float* __restrict__ kraw, const float* __restrict__ vpre,
    const float* __restrict__ w2p,  const float* __restrict__ a2p,
    const float* __restrict__ v2p,  const float* __restrict__ v_first,
    const float* __restrict__ w0, const float* __restrict__ a0,
    const float* __restrict__ v0, const float* __restrict__ k_k,
    const float* __restrict__ k_a,
    const unsigned* __restrict__ scal,             // [1]=max|v2p|  [2]=max|vd|
    float* __restrict__ wwout, float* __restrict__ kfin,
    float* __restrict__ vfin,  float* __restrict__ kkn,
    float* __restrict__ asig) {
  const int gw   = (int)((blockIdx.x * 256LL + threadIdx.x) >> 5);  // head-row id
  const int lane = threadIdx.x & 31;
  const int h    = gw & (Hh - 1);
  const long long rowbase = (long long)gw * Nh;
  const float sv2 = fmaxf(__uint_as_float(scal[1]), 1.f);
  const float svd = fmaxf(__uint_as_float(scal[2]), 1.f);
  float kkv[2]; long long idx[2];
  float sumsq = 0.f;
#pragma unroll
  for (int u = 0; u < 2; ++u) {
    const int n = lane + u * 32;
    const long long i = rowbase + n;
    idx[u] = i;
    const int c = h * Nh + n;
    const float kr  = kraw[i];
    const float kkx = kr * k_k[c];
    kkv[u] = kkx;
    sumsq += kkx * kkx;
    // w = -softplus(-(w0 + w2p)) - 0.5 ; store decay exp(-exp(w))
    const float wr = w0[c] + w2p[i];
    const float wfin = -softplusf_(-wr) - 0.5f;
    wwout[i] = expf(-expf(wfin));
    const float a = sigmoidf_(a0[c] + a2p[i]);
    asig[i] = a;
    kfin[i] = kr * (1.f + (a - 1.f) * k_a[c]);
    float gi = v0[c] + v2p[i] / sv2;
    gi = fminf(fmaxf(gi, -16.f), 16.f);
    const float gate = sigmoidf_(gi);
    const float vp = vpre[i];
    const float vd = (v_first[i] - vp) / svd;
    vfin[i] = vp + vd * gate;
  }
  for (int off = 16; off >= 1; off >>= 1) sumsq += __shfl_xor(sumsq, off, 32);
  const float inv = 1.f / fmaxf(sqrtf(sumsq), 1e-12f);
#pragma unroll
  for (int u = 0; u < 2; ++u) kkn[idx[u]] = kkv[u] * inv;
}

// ---------------- wkv7 recurrence: one block per (b,h), state in registers ----------------
// 256 threads: thread = (row i 0..63) x (col group jg 0..3 of 16).  S: 16 f32/thread.
// sa/y reductions are 4-lane shfl_xor -> no barriers at all.
__global__ __launch_bounds__(256) void wkv7_kernel(
    const float* __restrict__ r, const float* __restrict__ ww,
    const float* __restrict__ k, const float* __restrict__ v,
    const float* __restrict__ kkn, const float* __restrict__ asig,
    float* __restrict__ y) {
  const int bh = blockIdx.x;           // b*H + h
  const int b = bh >> 5, h = bh & 31;
  const int tid = threadIdx.x;
  const int i  = tid >> 2;
  const int jg = tid & 3;
  const int j0 = jg << 4;
  float S[16];
#pragma unroll
  for (int m = 0; m < 16; ++m) S[m] = 0.f;
  const long long base = (long long)b * Tv * Cv + h * Nh;

  for (int t = 0; t < Tv; ++t) {
    const long long o = base + (long long)t * Cv;
    float wv[16], kv[16], kk2[16], av[16], rv[16];
    const float4* wp  = (const float4*)(ww   + o + j0);
    const float4* kp  = (const float4*)(k    + o + j0);
    const float4* kkp = (const float4*)(kkn  + o + j0);
    const float4* ap  = (const float4*)(asig + o + j0);
    const float4* rp  = (const float4*)(r    + o + j0);
#pragma unroll
    for (int q = 0; q < 4; ++q) {
      ((float4*)wv)[q]  = wp[q];
      ((float4*)kv)[q]  = kp[q];
      ((float4*)kk2)[q] = kkp[q];
      ((float4*)av)[q]  = ap[q];
      ((float4*)rv)[q]  = rp[q];
    }
    const float vi = v[o + i];
    float s1 = 0.f;
#pragma unroll
    for (int m = 0; m < 16; ++m) s1 += S[m] * kk2[m];
    s1 += __shfl_xor(s1, 1, 32);
    s1 += __shfl_xor(s1, 2, 32);
    const float sa = -s1;                       // S @ (-kk)
    float yv = 0.f;
#pragma unroll
    for (int m = 0; m < 16; ++m) {
      const float bj = kk2[m] * av[m];          // b = kk * a
      S[m] = S[m] * wv[m] + sa * bj + vi * kv[m];
      yv += S[m] * rv[m];
    }
    yv += __shfl_xor(yv, 1, 32);
    yv += __shfl_xor(yv, 2, 32);
    if (jg == 0) y[o + i] = yv;
  }
}

// ---------------- GroupNorm + bonus + gate, emit bf16 for final GEMM ----------------
__global__ __launch_bounds__(256) void gnorm_kernel(
    const float* __restrict__ y, const float* __restrict__ r,
    const float* __restrict__ kfin, const float* __restrict__ vfin,
    const float* __restrict__ g, const float* __restrict__ r_k,
    const float* __restrict__ ln_w, const float* __restrict__ ln_b,
    __bf16* __restrict__ yg) {
  const int gw   = (int)((blockIdx.x * 256LL + threadIdx.x) >> 5);
  const int lane = threadIdx.x & 31;
  const int h    = gw & (Hh - 1);
  const long long rowbase = (long long)gw * Nh;
  float yv[2]; long long idx[2]; int cc[2];
  float sum = 0.f, sumsq = 0.f, dot = 0.f;
#pragma unroll
  for (int u = 0; u < 2; ++u) {
    const int n = lane + u * 32;
    const long long i = rowbase + n;
    idx[u] = i;
    cc[u] = h * Nh + n;
    const float yy = y[i];
    yv[u] = yy;
    sum += yy; sumsq += yy * yy;
    dot += r[i] * kfin[i] * r_k[h * Nh + n];
  }
  for (int off = 16; off >= 1; off >>= 1) {
    sum   += __shfl_xor(sum,   off, 32);
    sumsq += __shfl_xor(sumsq, off, 32);
    dot   += __shfl_xor(dot,   off, 32);
  }
  const float mu  = sum * (1.f / 64.f);
  const float var = sumsq * (1.f / 64.f) - mu * mu;
  const float inv = rsqrtf(var + EPSGN);
#pragma unroll
  for (int u = 0; u < 2; ++u) {
    const long long i = idx[u];
    const int c = cc[u];
    const float normed = (yv[u] - mu) * inv * ln_w[c] + ln_b[c];
    yg[i] = f2bf((normed + dot * vfin[i]) * g[i]);
  }
}

// ---------------- final rescale-by-0.5 (if max|y| > thr) + v_first passthrough ----------------
__global__ __launch_bounds__(256) void finalize_kernel(
    float* __restrict__ yout, const float* __restrict__ v_first,
    float* __restrict__ vout, const unsigned* __restrict__ ymax_bits) {
  long long idx = blockIdx.x * 256LL + threadIdx.x;
  if (idx >= BTC) return;
  const float f = (__uint_as_float(*ymax_bits) > RESCALE_THR) ? 0.5f : 1.0f;
  yout[idx] *= f;
  vout[idx] = v_first[idx] * f;
}

// ---------------- host side ----------------
static void launch_gemm(const __bf16* A, const __bf16* Bt, float* C,
                        int M, int N, int K, hipStream_t s) {
  dim3 grid((unsigned)((N + 63) / 64), (unsigned)(M / 128));
  gemm_bf16_kernel<<<grid, 256, 0, s>>>(A, Bt, C, M, N, K);
}

extern "C" void kernel_launch(void* const* d_in, const int* in_sizes, int n_in,
                              void* d_out, int out_size, void* d_ws, size_t ws_size,
                              hipStream_t stream) {
  (void)in_sizes; (void)n_in; (void)out_size; (void)ws_size;
  const float* x       = (const float*)d_in[0];
  const float* v_first = (const float*)d_in[1];
  const float* x_r = (const float*)d_in[2];
  const float* x_w = (const float*)d_in[3];
  const float* x_k = (const float*)d_in[4];
  const float* x_v = (const float*)d_in[5];
  const float* x_a = (const float*)d_in[6];
  const float* x_g = (const float*)d_in[7];
  const float* w0  = (const float*)d_in[8];
  const float* w1  = (const float*)d_in[9];
  const float* w2  = (const float*)d_in[10];
  const float* a0  = (const float*)d_in[11];
  const float* a1  = (const float*)d_in[12];
  const float* a2  = (const float*)d_in[13];
  const float* v0  = (const float*)d_in[14];
  const float* v1  = (const float*)d_in[15];
  const float* v2  = (const float*)d_in[16];
  const float* g1  = (const float*)d_in[17];
  const float* g2  = (const float*)d_in[18];
  const float* k_k = (const float*)d_in[19];
  const float* k_a = (const float*)d_in[20];
  const float* r_k = (const float*)d_in[21];
  const float* W_r = (const float*)d_in[22];
  const float* W_k = (const float*)d_in[23];
  const float* W_v = (const float*)d_in[24];
  const float* W_o = (const float*)d_in[25];
  const float* ln_w = (const float*)d_in[26];
  const float* ln_b = (const float*)d_in[27];

  // ---- workspace layout ----
  char* wsb = (char*)d_ws;
  size_t off = 0;
  auto alloc = [&](size_t bytes) -> void* {
    void* p = wsb + off;
    off += (bytes + 255) & ~(size_t)255;
    return p;
  };
  unsigned* scal = (unsigned*)alloc(256);   // [0]=max|v1p| [1]=max|v2p| [2]=max|vd| [3]=max|y|
  const size_t BTC2 = (size_t)BTC * 2, BTC4 = (size_t)BTC * 4;
  __bf16* xr_bf = (__bf16*)alloc(BTC2);
  __bf16* xw_bf = (__bf16*)alloc(BTC2);
  __bf16* xk_bf = (__bf16*)alloc(BTC2);
  __bf16* xv_bf = (__bf16*)alloc(BTC2);
  __bf16* xa_bf = (__bf16*)alloc(BTC2);
  __bf16* xg_bf = (__bf16*)alloc(BTC2);
  __bf16* Wr_bf = (__bf16*)alloc((size_t)Cv * Cv * 2);
  __bf16* Wk_bf = (__bf16*)alloc((size_t)Cv * Cv * 2);
  __bf16* Wv_bf = (__bf16*)alloc((size_t)Cv * Cv * 2);
  __bf16* Wo_bf = (__bf16*)alloc((size_t)Cv * Cv * 2);
  __bf16* w1t = (__bf16*)alloc((size_t)64  * Cv * 2);   // [64,2048]
  __bf16* w2t = (__bf16*)alloc((size_t)Cv  * 64 * 2);   // [2048,64]
  __bf16* a1t = (__bf16*)alloc((size_t)64  * Cv * 2);
  __bf16* a2t = (__bf16*)alloc((size_t)Cv  * 64 * 2);
  __bf16* v1t = (__bf16*)alloc((size_t)32  * Cv * 2);   // [32,2048]
  __bf16* v2t = (__bf16*)alloc((size_t)Cv  * 32 * 2);   // [2048,32]
  __bf16* g1t = (__bf16*)alloc((size_t)128 * Cv * 2);
  __bf16* g2t = (__bf16*)alloc((size_t)Cv  * 128 * 2);
  __bf16* hw_bf = (__bf16*)alloc((size_t)BT * 64 * 2);
  __bf16* ha_bf = (__bf16*)alloc((size_t)BT * 64 * 2);
  __bf16* hg_bf = (__bf16*)alloc((size_t)BT * 128 * 2);
  __bf16* v1b   = (__bf16*)alloc((size_t)BT * 32 * 2);
  __bf16* yg_bf = (__bf16*)alloc(BTC2);

  float* r_f    = (float*)alloc(BTC4);
  float* kraw_f = (float*)alloc(BTC4);
  float* vpre_f = (float*)alloc(BTC4);
  float* w2p_f  = (float*)alloc(BTC4);
  float* a2p_f  = (float*)alloc(BTC4);
  float* g_f    = (float*)alloc(BTC4);
  float* v2p_f  = (float*)alloc(BTC4);
  float* ww_f   = (float*)alloc(BTC4);
  float* kfin_f = (float*)alloc(BTC4);
  float* vfin_f = (float*)alloc(BTC4);
  float* kkn_f  = (float*)alloc(BTC4);
  float* asig_f = (float*)alloc(BTC4);
  float* y_f    = (float*)alloc(BTC4);
  float* hw_raw = (float*)alloc((size_t)BT * 64 * 4);
  float* ha_raw = (float*)alloc((size_t)BT * 64 * 4);
  float* hg_raw = (float*)alloc((size_t)BT * 128 * 4);
  float* v1p_f  = (float*)alloc((size_t)BT * 32 * 4);

  float* y_out = (float*)d_out;
  float* v_out = y_out + BTC;

  hipMemsetAsync(scal, 0, 256, stream);

  // ---- weights -> bf16 (small LoRA weights transposed to [N,K]) ----
  auto conv = [&](const float* s_, __bf16* d_, int rows_, int cols_, int tr) {
    long long n = (long long)rows_ * cols_;
    convert_bf16_kernel<<<(unsigned)((n + 255) / 256), 256, 0, stream>>>(s_, d_, rows_, cols_, tr);
  };
  conv(W_r, Wr_bf, Cv, Cv, 0);
  conv(W_k, Wk_bf, Cv, Cv, 0);
  conv(W_v, Wv_bf, Cv, Cv, 0);
  conv(W_o, Wo_bf, Cv, Cv, 0);
  conv(w1, w1t, Cv, 64, 1);   conv(w2, w2t, 64, Cv, 1);
  conv(a1, a1t, Cv, 64, 1);   conv(a2, a2t, 64, Cv, 1);
  conv(v1, v1t, Cv, 32, 1);   conv(v2, v2t, 32, Cv, 1);
  conv(g1, g1t, Cv, 128, 1);  conv(g2, g2t, 128, Cv, 1);

  // ---- token shift + mixes ----
  prep_mix_kernel<<<(unsigned)(BTC / 256), 256, 0, stream>>>(
      x, x_r, x_w, x_k, x_v, x_a, x_g,
      xr_bf, xw_bf, xk_bf, xv_bf, xa_bf, xg_bf);

  // ---- big + first-stage LoRA GEMMs ----
  launch_gemm(xr_bf, Wr_bf, r_f,    BT, Cv, Cv, stream);
  launch_gemm(xk_bf, Wk_bf, kraw_f, BT, Cv, Cv, stream);
  launch_gemm(xv_bf, Wv_bf, vpre_f, BT, Cv, Cv, stream);
  launch_gemm(xw_bf, w1t, hw_raw, BT, 64,  Cv, stream);
  launch_gemm(xa_bf, a1t, ha_raw, BT, 64,  Cv, stream);
  launch_gemm(xg_bf, g1t, hg_raw, BT, 128, Cv, stream);
  launch_gemm(xv_bf, v1t, v1p_f,  BT, 32,  Cv, stream);

  // ---- mid activations, second-stage LoRA GEMMs ----
  act_bf16_kernel<<<(unsigned)((BT * 64 + 255) / 256), 256, 0, stream>>>(hw_raw, hw_bf, BT * 64, 1, scal);
  act_bf16_kernel<<<(unsigned)((BT * 64 + 255) / 256), 256, 0, stream>>>(ha_raw, ha_bf, BT * 64, 0, scal);
  act_bf16_kernel<<<(unsigned)((BT * 128 + 255) / 256), 256, 0, stream>>>(hg_raw, hg_bf, BT * 128, 2, scal);
  launch_gemm(hw_bf, w2t, w2p_f, BT, Cv, 64,  stream);
  launch_gemm(ha_bf, a2t, a2p_f, BT, Cv, 64,  stream);
  launch_gemm(hg_bf, g2t, g_f,   BT, Cv, 128, stream);

  // ---- value-residual path: v1p -> (maxabs,scale) -> @v2 -> maxabs ----
  maxabs_kernel<<<1024, 256, 0, stream>>>(v1p_f, (const float*)nullptr, (long long)BT * 32, scal + 0);
  act_bf16_kernel<<<(unsigned)((BT * 32 + 255) / 256), 256, 0, stream>>>(v1p_f, v1b, BT * 32, 3, scal + 0);
  launch_gemm(v1b, v2t, v2p_f, BT, Cv, 32, stream);
  maxabs_kernel<<<1024, 256, 0, stream>>>(v2p_f, (const float*)nullptr, BTC, scal + 1);
  maxabs_kernel<<<1024, 256, 0, stream>>>(v_first, vpre_f, BTC, scal + 2);   // max|v_first - v|

  // ---- fused elementwise (decay, a, k_final, gate, kk-normalize) ----
  post_kernel<<<8192, 256, 0, stream>>>(
      kraw_f, vpre_f, w2p_f, a2p_f, v2p_f, v_first,
      w0, a0, v0, k_k, k_a, scal,
      ww_f, kfin_f, vfin_f, kkn_f, asig_f);

  // ---- sequential wkv7 scan ----
  wkv7_kernel<<<Bv * Hh, 256, 0, stream>>>(r_f, ww_f, kfin_f, vfin_f, kkn_f, asig_f, y_f);

  maxabs_kernel<<<1024, 256, 0, stream>>>(y_f, (const float*)nullptr, BTC, scal + 3);

  // ---- groupnorm + bonus + gate -> bf16 ----
  gnorm_kernel<<<8192, 256, 0, stream>>>(
      y_f, r_f, kfin_f, vfin_f, g_f, r_k, ln_w, ln_b, yg_bf);

  // ---- output projection + conditional 0.5 rescale ----
  launch_gemm(yg_bf, Wo_bf, y_out, BT, Cv, Cv, stream);
  finalize_kernel<<<(unsigned)(BTC / 256), 256, 0, stream>>>(y_out, v_first, v_out, scal + 3);
}
